// NonLocalBlock_71528385348137
// MI455X (gfx1250) — matrix operand order repaired and checked by reading, not compiled, not run on hardware
//
#include <hip/hip_runtime.h>
#include <hip/hip_bf16.h>

// ---------------------------------------------------------------------------
// NonLocalBlock on MI455X (gfx1250): flash-attention over spatial positions
// using v_wmma_f32_16x16x32_f16 + double-buffered GLOBAL_LOAD_ASYNC_TO_LDS
// staging of K/V tiles (ASYNCcnt / s_wait_asynccnt). wave32 layouts per
// CDNA5 ISA 7.12.2.
// ---------------------------------------------------------------------------

typedef _Float16 f16;
typedef __attribute__((ext_vector_type(16))) _Float16 v16h;
typedef __attribute__((ext_vector_type(8)))  _Float16 v8h;
typedef __attribute__((ext_vector_type(8)))  float    v8f;

#define BATCH   8
#define C_IN    256
#define C_INT   128
#define NPOS    4096   // 64*64

__device__ __forceinline__ v16h make_v16(v8h lo, v8h hi) {
    v16h r;
#pragma unroll
    for (int i = 0; i < 8; ++i) { r[i] = lo[i]; r[i + 8] = hi[i]; }
    return r;
}

// A-matrix fragment (16x32 f16). 'row_ptr' points at this lane's row.
// Lane (half = lane>>4) holds k in [8h, 8h+7] and [16+8h, 16+8h+7].
__device__ __forceinline__ v16h load_a_frag(const f16* row_ptr, int half) {
    v8h lo = *(const v8h*)(row_ptr + 8 * half);
    v8h hi = *(const v8h*)(row_ptr + 16 + 8 * half);
    return make_v16(lo, hi);
}

// B-matrix fragment (32x16 f16). Buffer stores the matrix COLUMN contiguously:
// lane (half = lane>>4) holds k in [8h, 8h+7] and [16+8h, 16+8h+7].
__device__ __forceinline__ v16h load_b_frag(const f16* col_ptr, int half) {
    const f16* p = col_ptr + 8 * half;
    return make_v16(*(const v8h*)p, *(const v8h*)(p + 16));
}

__device__ __forceinline__ v8f wmma_f16(v16h a, v16h b, v8f c) {
    return __builtin_amdgcn_wmma_f32_16x16x32_f16(false, a, false, b,
                                                  (short)0, c, false, false);
}

// Async DMA: global -> LDS, 16 bytes per lane, tracked by ASYNCcnt.
__device__ __forceinline__ void async_b128(unsigned lds_off, const f16* gaddr) {
    asm volatile("global_load_async_to_lds_b128 %0, %1, off"
                 :: "v"(lds_off), "v"(gaddr) : "memory");
}
__device__ __forceinline__ unsigned lds_off(const void* p) {
    return (unsigned)(uintptr_t)p;   // low 32 bits of generic ptr = LDS offset
}

// ---------------------------------------------------------------------------
// Stage 0a: x (B,C,N) f32  ->  x_t (B,N,C) f16   (tiled LDS transpose)
// ---------------------------------------------------------------------------
__global__ void k_transpose_x(const float* __restrict__ x, f16* __restrict__ xt) {
    __shared__ float tile[32][33];
    const int b  = blockIdx.z;
    const int n0 = blockIdx.x * 32;
    const int c0 = blockIdx.y * 32;
    const float* xb = x + (size_t)b * C_IN * NPOS;
    f16* xtb = xt + (size_t)b * NPOS * C_IN;
#pragma unroll
    for (int i = threadIdx.y; i < 32; i += 8)
        tile[i][threadIdx.x] = xb[(size_t)(c0 + i) * NPOS + n0 + threadIdx.x];
    __syncthreads();
#pragma unroll
    for (int i = threadIdx.y; i < 32; i += 8)
        xtb[(size_t)(n0 + i) * C_IN + c0 + threadIdx.x] = (f16)tile[threadIdx.x][i];
}

// ---------------------------------------------------------------------------
// Stage 0b: f32 -> f16 weight conversion
// ---------------------------------------------------------------------------
__global__ void k_cvt_f16(const float* __restrict__ src, f16* __restrict__ dst, int n) {
    int i = blockIdx.x * blockDim.x + threadIdx.x;
    if (i < n) dst[i] = (f16)src[i];
}

// ---------------------------------------------------------------------------
// Stage 1: projections.  One wave per (b, 16-wide n tile, 16-wide i tile).
//   theta[b][n][i], phi[b][n][i]  (row-major over i, A/B-frag friendly)
//   gt[b][o][n]                    (transposed: key-contiguous for PV B-frags)
// ---------------------------------------------------------------------------
__global__ void k_proj(const f16* __restrict__ xt,
                       const f16* __restrict__ wgh, const f16* __restrict__ wth,
                       const f16* __restrict__ wph,
                       const float* __restrict__ bg, const float* __restrict__ bt,
                       const float* __restrict__ bp,
                       f16* __restrict__ theta, f16* __restrict__ phi,
                       f16* __restrict__ gt) {
    const int lane = threadIdx.x & 31;
    const int half = lane >> 4;
    const int mrow = lane & 15;
    const int n0 = blockIdx.x * 16;
    const int i0 = blockIdx.y * 16;
    const int b  = blockIdx.z;

    const f16* xb = xt + (size_t)b * NPOS * C_IN + (size_t)(n0 + mrow) * C_IN;

    v8f accg = {}, acct = {}, accp = {};
#pragma unroll
    for (int k0 = 0; k0 < C_IN; k0 += 32) {
        v16h bx = load_b_frag(xb + k0, half);                 // x_t as B (K=c, N=n)
        v16h ag = load_a_frag(wgh + (size_t)(i0 + mrow) * C_IN + k0, half);
        v16h at = load_a_frag(wth + (size_t)(i0 + mrow) * C_IN + k0, half);
        v16h ap = load_a_frag(wph + (size_t)(i0 + mrow) * C_IN + k0, half);
        accg = wmma_f16(ag, bx, accg);
        acct = wmma_f16(at, bx, acct);
        accp = wmma_f16(ap, bx, accp);
    }
    // D layout: col n = n0+mrow (fixed per lane), rows i = i0 + v + 8*half.
    v8f bgv = *(const v8f*)(bg + i0 + 8 * half);
    v8f btv = *(const v8f*)(bt + i0 + 8 * half);
    v8f bpv = *(const v8f*)(bp + i0 + 8 * half);

    v8h hg, ht, hp;
#pragma unroll
    for (int v = 0; v < 8; ++v) {
        hg[v] = (f16)(accg[v] + bgv[v]);
        ht[v] = (f16)(acct[v] + btv[v]);
        hp[v] = (f16)(accp[v] + bpv[v]);
    }
    const size_t nrow = (size_t)b * NPOS + n0 + mrow;
    *(v8h*)(theta + nrow * C_INT + i0 + 8 * half) = ht;
    *(v8h*)(phi   + nrow * C_INT + i0 + 8 * half) = hp;
    // g transposed: gt[b][o][n]
    f16* gb = gt + (size_t)b * C_INT * NPOS;
#pragma unroll
    for (int v = 0; v < 8; ++v)
        gb[(size_t)(i0 + 8 * half + v) * NPOS + n0 + mrow] = hg[v];
}

// ---------------------------------------------------------------------------
// Stage 2: flash attention. 8 waves/block, one 16-query tile each; all waves
// share double-buffered LDS K/V tiles filled by GLOBAL_LOAD_ASYNC_TO_LDS_B128.
// 16 WMMAs per wave per 32-key tile (8 QK + 8 PV). P converted C-layout ->
// A-layout through per-wave LDS staging.
// ---------------------------------------------------------------------------
#define WAVES 8
#define KROW  136   // 32 keys x 128 d, padded: 272B rows (16B aligned, bank-skewed)
#define VROW  40    // 128 o x 32 keys, padded: 80B rows (16B aligned, conflict-free)
#define PROW  48

// Issue one 32-key tile: K = phi[key0..key0+31][0..127], V = gt[0..127][key0..+31].
// 1024 x 16B chunks split across 256 threads -> 4 async ops per thread.
__device__ __forceinline__ void issue_tile(const f16* __restrict__ ph,
                                           const f16* __restrict__ gv,
                                           int key0, f16 (*kb)[KROW],
                                           f16 (*vb)[VROW], int tid) {
#pragma unroll
    for (int j = 0; j < 2; ++j) {        // K tile: 512 chunks, 16/row
        int c = tid + 256 * j;
        int row = c >> 4, col = c & 15;
        async_b128(lds_off(&kb[row][col * 8]),
                   ph + (size_t)(key0 + row) * C_INT + col * 8);
    }
#pragma unroll
    for (int j = 0; j < 2; ++j) {        // V tile: 512 chunks, 4/row
        int c = tid + 256 * j;
        int row = c >> 2, col = c & 3;
        async_b128(lds_off(&vb[row][col * 8]),
                   gv + (size_t)row * NPOS + key0 + col * 8);
    }
}

__global__ void k_attn(const f16* __restrict__ theta, const f16* __restrict__ phi,
                       const f16* __restrict__ gt, f16* __restrict__ y) {
    __shared__ alignas(16) f16 Kb[2][32][KROW];
    __shared__ alignas(16) f16 Vb[2][128][VROW];
    __shared__ alignas(16) f16 Pb[WAVES][16][PROW];

    const int tid  = threadIdx.x;
    const int wid  = tid >> 5;
    const int lane = tid & 31;
    const int half = lane >> 4;
    const int mrow = lane & 15;
    const int b  = blockIdx.y;
    const int n0 = (blockIdx.x * WAVES + wid) * 16;

    const f16* th = theta + ((size_t)b * NPOS + n0 + mrow) * C_INT;
    const f16* ph = phi + (size_t)b * NPOS * C_INT;
    const f16* gv = gt  + (size_t)b * C_INT * NPOS;

    // Q fragments resident: 4 x (16x32) over d=128
    v16h qf[4];
#pragma unroll
    for (int i = 0; i < 4; ++i) qf[i] = load_a_frag(th + i * 32, half);

    v8f zero = {};
    v8f acc[8];
#pragma unroll
    for (int f = 0; f < 8; ++f) acc[f] = zero;
    float mrv[8], lrow[8];
#pragma unroll
    for (int v = 0; v < 8; ++v) { mrv[v] = -1e30f; lrow[v] = 0.0f; }

    f16* pb = &Pb[wid][0][0];

    // Prologue: DMA tile 0 into buffer 0.
    issue_tile(ph, gv, 0, Kb[0], Vb[0], tid);

    for (int key0 = 0; key0 < NPOS; key0 += 32) {
        const int cb = (key0 >> 5) & 1;
        if (key0 + 32 < NPOS) {
            issue_tile(ph, gv, key0 + 32, Kb[cb ^ 1], Vb[cb ^ 1], tid);
            asm volatile("s_wait_asynccnt 0x4" ::: "memory");  // tile key0 done
        } else {
            asm volatile("s_wait_asynccnt 0x0" ::: "memory");
        }
        __syncthreads();   // all waves' DMA for this tile visible

        // ---- scores: S[16q x 32k] as two 16x16 f32 accumulators (K from LDS)
        v8f s0 = zero, s1 = zero;
#pragma unroll
        for (int kd = 0; kd < 4; ++kd) {
            s0 = wmma_f16(qf[kd], load_b_frag(&Kb[cb][mrow][kd * 32], half), s0);
            s1 = wmma_f16(qf[kd], load_b_frag(&Kb[cb][16 + mrow][kd * 32], half), s1);
        }
        // ---- online softmax: rows live at (v + 8*half), cols across 16 lanes
        float mnew[8], scale[8];
#pragma unroll
        for (int v = 0; v < 8; ++v) {
            float t = fmaxf(s0[v], s1[v]);
#pragma unroll
            for (int off = 1; off < 16; off <<= 1)
                t = fmaxf(t, __shfl_xor(t, off, 32));
            mnew[v]  = fmaxf(mrv[v], t);
            scale[v] = __expf(mrv[v] - mnew[v]);
            mrv[v]   = mnew[v];
        }
#pragma unroll
        for (int v = 0; v < 8; ++v) {
            float p0 = __expf(s0[v] - mnew[v]);
            float p1 = __expf(s1[v] - mnew[v]);
            pb[(v + 8 * half) * PROW + mrow]      = (f16)p0;   // C-layout -> LDS
            pb[(v + 8 * half) * PROW + 16 + mrow] = (f16)p1;
            float t = p0 + p1;
#pragma unroll
            for (int off = 1; off < 16; off <<= 1)
                t += __shfl_xor(t, off, 32);
            lrow[v] = lrow[v] * scale[v] + t;
        }
#pragma unroll
        for (int f = 0; f < 8; ++f)
#pragma unroll
            for (int v = 0; v < 8; ++v) acc[f][v] *= scale[v];

        // ---- reload P as an A-fragment (16x32) from LDS
        v16h pf = load_a_frag(pb + mrow * PROW, half);

        // ---- PV: 8 output-channel tiles, K = 32 keys (V from LDS)
#pragma unroll
        for (int f = 0; f < 8; ++f)
            acc[f] = wmma_f16(pf, load_b_frag(&Vb[cb][f * 16 + mrow][0], half),
                              acc[f]);

        __syncthreads();   // done reading this buffer before it is re-filled
    }

    float rinv[8];
#pragma unroll
    for (int v = 0; v < 8; ++v) rinv[v] = 1.0f / lrow[v];

    f16* yb = y + ((size_t)b * NPOS + n0) * C_INT;
#pragma unroll
    for (int f = 0; f < 8; ++f)
#pragma unroll
        for (int v = 0; v < 8; ++v)
            yb[(size_t)(v + 8 * half) * C_INT + f * 16 + mrow] =
                (f16)(acc[f][v] * rinv[v]);
}

// ---------------------------------------------------------------------------
// Stage 3: out[b][c][n] = wW[c,:] . y[b][n][:] + bW[c] + x[b][c][n]
// One wave per (b, 16-n tile, 64-c slab): 16 WMMAs.
// ---------------------------------------------------------------------------
__global__ void k_out(const f16* __restrict__ y, const f16* __restrict__ wWh,
                      const float* __restrict__ bW, const float* __restrict__ x,
                      float* __restrict__ out) {
    const int lane = threadIdx.x & 31;
    const int half = lane >> 4;
    const int mrow = lane & 15;
    const int n0 = blockIdx.x * 16;
    const int c0 = blockIdx.y * 64;
    const int b  = blockIdx.z;

    const f16* yrow = y + ((size_t)b * NPOS + n0 + mrow) * C_INT;
    v16h af[4];
#pragma unroll
    for (int i = 0; i < 4; ++i) af[i] = load_a_frag(yrow + i * 32, half);

    v8f zero = {};
    v8f acc[4];
#pragma unroll
    for (int ct = 0; ct < 4; ++ct) acc[ct] = zero;

#pragma unroll
    for (int kd = 0; kd < 4; ++kd)
#pragma unroll
        for (int ct = 0; ct < 4; ++ct) {
            const f16* pw = wWh + (size_t)(c0 + ct * 16 + mrow) * C_INT + kd * 32;
            acc[ct] = wmma_f16(af[kd], load_b_frag(pw, half), acc[ct]);
        }

    const float* xb = x + (size_t)b * C_IN * NPOS;
    float* ob = out + (size_t)b * C_IN * NPOS;
#pragma unroll
    for (int ct = 0; ct < 4; ++ct) {
        const int c = c0 + ct * 16 + mrow;           // D col per lane
        const float bias = bW[c];
        const size_t base = (size_t)c * NPOS + n0 + 8 * half;
        v8f xv = *(const v8f*)(xb + base);
        v8f r;
#pragma unroll
        for (int v = 0; v < 8; ++v) r[v] = acc[ct][v] + bias + xv[v];
        *(v8f*)(ob + base) = r;
    }
}

// ---------------------------------------------------------------------------
extern "C" void kernel_launch(void* const* d_in, const int* in_sizes, int n_in,
                              void* d_out, int out_size, void* d_ws, size_t ws_size,
                              hipStream_t stream) {
    const float* x  = (const float*)d_in[0];
    const float* wg = (const float*)d_in[1];
    const float* bg = (const float*)d_in[2];
    const float* wt = (const float*)d_in[3];
    const float* bt = (const float*)d_in[4];
    const float* wp = (const float*)d_in[5];
    const float* bp = (const float*)d_in[6];
    const float* wW = (const float*)d_in[7];
    const float* bW = (const float*)d_in[8];
    float* out = (float*)d_out;

    char* ws = (char*)d_ws;
    size_t off = 0;
    auto alloc = [&](size_t bytes) -> void* {
        void* p = ws + off;
        off += (bytes + 255) & ~(size_t)255;
        return p;
    };
    f16* xt    = (f16*)alloc((size_t)BATCH * NPOS * C_IN * sizeof(f16));   // 16 MB
    f16* wgh   = (f16*)alloc((size_t)C_INT * C_IN * sizeof(f16));
    f16* wth   = (f16*)alloc((size_t)C_INT * C_IN * sizeof(f16));
    f16* wph   = (f16*)alloc((size_t)C_INT * C_IN * sizeof(f16));
    f16* wWh   = (f16*)alloc((size_t)C_IN * C_INT * sizeof(f16));
    f16* theta = (f16*)alloc((size_t)BATCH * NPOS * C_INT * sizeof(f16));  // 8 MB
    f16* phi   = (f16*)alloc((size_t)BATCH * NPOS * C_INT * sizeof(f16));  // 8 MB
    f16* gt    = (f16*)alloc((size_t)BATCH * NPOS * C_INT * sizeof(f16));  // 8 MB
    f16* yws   = (f16*)alloc((size_t)BATCH * NPOS * C_INT * sizeof(f16));  // 8 MB

    // Stage 0: conversions
    {
        int nw = C_INT * C_IN;
        dim3 blk(256), grd((nw + 255) / 256);
        k_cvt_f16<<<grd, blk, 0, stream>>>(wg, wgh, nw);
        k_cvt_f16<<<grd, blk, 0, stream>>>(wt, wth, nw);
        k_cvt_f16<<<grd, blk, 0, stream>>>(wp, wph, nw);
        k_cvt_f16<<<grd, blk, 0, stream>>>(wW, wWh, nw);
    }
    k_transpose_x<<<dim3(NPOS / 32, C_IN / 32, BATCH), dim3(32, 8), 0, stream>>>(x, xt);

    // Stage 1: projections (one wave per block)
    k_proj<<<dim3(NPOS / 16, C_INT / 16, BATCH), dim3(32), 0, stream>>>(
        xt, wgh, wth, wph, bg, bt, bp, theta, phi, gt);

    // Stage 2: flash attention (8 waves/block, async double-buffered K/V)
    k_attn<<<dim3(NPOS / (16 * WAVES), BATCH), dim3(32 * WAVES), 0, stream>>>(
        theta, phi, gt, yws);

    // Stage 3: output projection + bias + residual
    k_out<<<dim3(NPOS / 16, C_IN / 64, BATCH), dim3(32), 0, stream>>>(
        yws, wWh, bW, x, out);
}
